// StackedLSTMs_60773787238728
// MI455X (gfx1250) — matrix operand order repaired and checked by reading, not compiled, not run on hardware
//
#include <hip/hip_runtime.h>
#include <hip/hip_bf16.h>
#include <stdint.h>

// ---------------- CDNA5 WMMA types ----------------
typedef __attribute__((ext_vector_type(16))) __bf16 v16bf;
typedef __attribute__((ext_vector_type(8)))  float  v8f;

union Frag16 {            // one wave-lane's 16-bf16 WMMA operand (8 VGPRs)
    v16bf    v;
    uint32_t u[8];
};

// round-to-nearest-even f32 -> bf16, packed pair into one dword
__device__ __forceinline__ uint32_t pack_bf16(float a, float b) {
    uint32_t ua = __builtin_bit_cast(uint32_t, a);
    uint32_t ub = __builtin_bit_cast(uint32_t, b);
    ua += 0x7FFFu + ((ua >> 16) & 1u);
    ub += 0x7FFFu + ((ub >> 16) & 1u);
    return (ua >> 16) | (ub & 0xFFFF0000u);
}

__device__ __forceinline__ uint16_t f32_to_bf16(float a) {
    uint32_t ua = __builtin_bit_cast(uint32_t, a);
    ua += 0x7FFFu + ((ua >> 16) & 1u);
    return (uint16_t)(ua >> 16);
}

__device__ __forceinline__ float sigm(float x) {
    return 1.0f / (1.0f + __expf(-x));
}

// =====================================================================
// Kernel 1: fused LSTM cell.
//   gates = [inputs|hx] @ [W_ih|W_hh]^T + b_ih + b_hh   (K = 256)
//   c' = sig(f)*cx + sig(i)*tanh(g);  h' = sig(o)*tanh(c')
// Block: 256 threads (8 waves). blockIdx.y picks 16 hidden units
// (=> 64 gate columns: the same 16 units of i,f,g,o). Grid-stride over
// 32-row M tiles. Waves: (rowTile 0..1) x (gate 0..3), each does a
// 16x16 C tile over K=256 via 8x v_wmma_f32_16x16x32_bf16.
// h' stored to workspace as bf16 (halves gather traffic for kernel 2).
// =====================================================================
__global__ __launch_bounds__(256)
void lstm_cell_kernel(const float* __restrict__ inputs,
                      const float* __restrict__ hx,
                      const float* __restrict__ cx,
                      const float* __restrict__ W_ih,
                      const float* __restrict__ W_hh,
                      const float* __restrict__ b_ih,
                      const float* __restrict__ b_hh,
                      uint16_t* __restrict__ h_bf,
                      int num_teams)
{
    // B fragments: [gate 0..3][kstep 0..7][lane 0..31][dword 0..7]  (32 KB)
    __shared__ uint32_t B_lds[4][8][32][8];
    // A fragments: [rowTile 0..1][kstep 0..7][lane 0..31][dword 0..7] (16 KB)
    __shared__ uint32_t A_lds[2][8][32][8];
    // gates staging: [m_local][gate][n] padded                      (~8.5 KB)
    __shared__ float gates_lds[32][4][17];
    __shared__ float bsum[4][16];

    const int tid  = threadIdx.x;
    const int lane = tid & 31;
    const int wv   = tid >> 5;          // 0..7
    const int j0   = blockIdx.y * 16;   // hidden-unit base for this block

    // ---- fill B fragments once per block (f32 -> bf16) ----
    // layout (16-bit B 32x16): lanes 0-15 hold K=0..7, 16..23 (2/VGPR),
    //                          lanes 16-31 hold K=8..15, 24..31
    for (int e = tid; e < 4 * 8 * 32 * 8; e += 256) {
        int i  = e & 7;
        int l  = (e >> 3) & 31;
        int s  = (e >> 8) & 7;
        int ct = (e >> 11) & 3;         // gate index
        int n    = l & 15;
        int grow = ct * 128 + j0 + n;   // row of [W_ih|W_hh] (gate output)
        int K    = 32 * s + ((l < 16) ? 0 : 8) + 2 * i + ((i >= 4) ? 8 : 0);
        float w0, w1;
        if (K < 128) { w0 = W_ih[grow * 128 + K];       w1 = W_ih[grow * 128 + K + 1]; }
        else         { w0 = W_hh[grow * 128 + K - 128]; w1 = W_hh[grow * 128 + K - 127]; }
        B_lds[ct][s][l][i] = pack_bf16(w0, w1);
    }
    if (tid < 64) {
        int ct = tid >> 4, n = tid & 15;
        int grow = ct * 128 + j0 + n;
        bsum[ct][n] = b_ih[grow] + b_hh[grow];
    }
    __syncthreads();

    const int ntiles = num_teams >> 5;  // 100000/32 = 3125, exact
    for (int mt = blockIdx.x; mt < ntiles; mt += gridDim.x) {
        const int m0 = mt << 5;

        // prefetch next M tile for this block (-> global_prefetch_b8)
        {
            int mn = mt + gridDim.x;
            if (mn < ntiles) {
                __builtin_prefetch(inputs + (size_t)mn * 32 * 128, 0, 0);
                __builtin_prefetch(hx     + (size_t)mn * 32 * 128, 0, 0);
            }
        }

        // ---- stage A tile (32 rows x K=256) as bf16 fragments ----
        // A layout (16-bit A 16x32): lanes 0-15: K=0..7 (VGPR0-3), 16..23 (4-7)
        //                            lanes 16-31: K=8..15, 24..31
        for (int e = tid; e < 32 * 128; e += 256) {
            int kp  = e & 127;          // K-pair index (K = 2*kp)
            int ml  = e >> 7;           // local row 0..31
            int K   = kp << 1;
            int row = m0 + ml;
            float a0, a1;
            if (K < 128) { a0 = inputs[row * 128 + K];       a1 = inputs[row * 128 + K + 1]; }
            else         { a0 = hx[row * 128 + K - 128];     a1 = hx[row * 128 + K - 127]; }
            int s    = K >> 5;
            int r32  = K & 31;
            int g16  = r32 >> 4;
            int pos8 = r32 & 15;
            int l    = (ml & 15) + ((pos8 >= 8) ? 16 : 0);
            int i    = g16 * 4 + ((pos8 & 7) >> 1);
            A_lds[ml >> 4][s][l][i] = pack_bf16(a0, a1);
        }
        __syncthreads();

        // ---- WMMA: wave (rt, ct) computes rows [rt*16, rt*16+16) of gate ct
        const int rt = wv >> 2;
        const int ct = wv & 3;
        v8f acc = {};
        #pragma unroll
        for (int s = 0; s < 8; ++s) {
            Frag16 a, b;
            #pragma unroll
            for (int i = 0; i < 8; ++i) {
                a.u[i] = A_lds[rt][s][lane][i];
                b.u[i] = B_lds[ct][s][lane][i];
            }
            acc = __builtin_amdgcn_wmma_f32_16x16x32_bf16(
                false, a.v, false, b.v, (short)0, acc, false, false);
        }

        // ---- scatter C tile to LDS (C layout: lane= (m>=8?16:0)+n, vgpr=m%8)
        {
            int n     = lane & 15;
            int mbase = rt * 16 + ((lane >= 16) ? 8 : 0);
            #pragma unroll
            for (int r = 0; r < 8; ++r)
                gates_lds[mbase + r][ct][n] = acc[r];
        }
        __syncthreads();

        // ---- fused LSTM elementwise + bf16 store of h' ----
        for (int e = tid; e < 32 * 16; e += 256) {
            int jl  = e & 15;
            int ml  = e >> 4;
            int row = m0 + ml;
            int j   = j0 + jl;
            float ig = gates_lds[ml][0][jl] + bsum[0][jl];
            float fg = gates_lds[ml][1][jl] + bsum[1][jl];
            float gg = gates_lds[ml][2][jl] + bsum[2][jl];
            float og = gates_lds[ml][3][jl] + bsum[3][jl];
            float c_new = sigm(fg) * cx[row * 128 + j] + sigm(ig) * tanhf(gg);
            float h_new = sigm(og) * tanhf(c_new);
            h_bf[row * 128 + j] = f32_to_bf16(h_new);
        }
        __syncthreads();   // protect gates_lds / A_lds for next iteration
    }
}

// =====================================================================
// Kernel 2: gather match pairs, project with W_out (3x256, zero-padded
// to 16 cols), softmax.  16 matches per wave, 8 waves per block.
// A fragment built directly from gathered bf16 h' rows (16B chunks map
// exactly onto the 16-bit A fragment layout). 8x wmma per wave.
// =====================================================================
__global__ __launch_bounds__(256)
void match_head_kernel(const uint16_t* __restrict__ h_bf,
                       const int* __restrict__ matches,
                       const float* __restrict__ W_out,
                       const float* __restrict__ b_out,
                       float* __restrict__ pmf,
                       int num_matches)
{
    __shared__ uint32_t Bw[8][32][8];   // W_out bf16 B-fragments (8 KB)
    __shared__ float    lg[8][16][4];   // per-wave logits (m, n<3)

    const int tid  = threadIdx.x;
    const int lane = tid & 31;
    const int wv   = tid >> 5;

    // ---- W_out -> bf16 B fragments (cols 3..15 are zero) ----
    for (int e = tid; e < 8 * 32 * 8; e += 256) {
        int i = e & 7;
        int l = (e >> 3) & 31;
        int s = (e >> 8) & 7;
        int n = l & 15;
        int K = 32 * s + ((l < 16) ? 0 : 8) + 2 * i + ((i >= 4) ? 8 : 0);
        float w0 = 0.0f, w1 = 0.0f;
        if (n < 3) { w0 = W_out[n * 256 + K]; w1 = W_out[n * 256 + K + 1]; }
        Bw[s][l][i] = pack_bf16(w0, w1);
    }
    __syncthreads();

    const int mbase = (blockIdx.x * 8 + wv) * 16;   // first match of this wave

    // Gather source rows for this lane's match (clamped so every lane —
    // and therefore WMMA with full EXEC — always does valid work).
    int lm   = lane & 15;
    int mid  = mbase + lm;
    int safe = (mid < num_matches) ? mid : (num_matches - 1);
    const uint16_t* hp = h_bf + (size_t)matches[2 * safe]     * 128;
    const uint16_t* ap = h_bf + (size_t)matches[2 * safe + 1] * 128;

    v8f acc = {};
    #pragma unroll
    for (int s = 0; s < 8; ++s) {
        const int kk   = 32 * s;
        const int off0 = kk + ((lane < 16) ? 0 : 8);        // -> u[0..3]
        const int off1 = kk + 16 + ((lane < 16) ? 0 : 8);   // -> u[4..7]
        Frag16 a, b;
        {   // each 8-element chunk lies fully inside home (K<128) or away
            const uint16_t* p0 = (off0 < 128) ? (hp + off0) : (ap + off0 - 128);
            const uint16_t* p1 = (off1 < 128) ? (hp + off1) : (ap + off1 - 128);
            uint4 t0 = *reinterpret_cast<const uint4*>(p0);
            uint4 t1 = *reinterpret_cast<const uint4*>(p1);
            a.u[0] = t0.x; a.u[1] = t0.y; a.u[2] = t0.z; a.u[3] = t0.w;
            a.u[4] = t1.x; a.u[5] = t1.y; a.u[6] = t1.z; a.u[7] = t1.w;
        }
        #pragma unroll
        for (int i = 0; i < 8; ++i) b.u[i] = Bw[s][lane][i];
        acc = __builtin_amdgcn_wmma_f32_16x16x32_bf16(
            false, a.v, false, b.v, (short)0, acc, false, false);
    }

    // ---- stash logits (C element (m,n): lane=(m>=8?16:0)+n, vgpr=m%8) ----
    {
        int n = lane & 15;
        if (n < 3) {
            int mb2 = (lane >= 16) ? 8 : 0;
            #pragma unroll
            for (int r = 0; r < 8; ++r)
                lg[wv][mb2 + r][n] = acc[r];
        }
    }
    __syncthreads();

    // ---- 3-way softmax, one match per lane (lanes 0..15) ----
    if (lane < 16) {
        int m = mbase + lane;
        if (m < num_matches) {
            float l0 = lg[wv][lane][0] + b_out[0];
            float l1 = lg[wv][lane][1] + b_out[1];
            float l2 = lg[wv][lane][2] + b_out[2];
            float mx = fmaxf(l0, fmaxf(l1, l2));
            float e0 = __expf(l0 - mx);
            float e1 = __expf(l1 - mx);
            float e2 = __expf(l2 - mx);
            float inv = 1.0f / (e0 + e1 + e2);
            pmf[3 * m + 0] = e0 * inv;
            pmf[3 * m + 1] = e1 * inv;
            pmf[3 * m + 2] = e2 * inv;
        }
    }
}

// =====================================================================
// Launch
// =====================================================================
extern "C" void kernel_launch(void* const* d_in, const int* in_sizes, int n_in,
                              void* d_out, int out_size, void* d_ws, size_t ws_size,
                              hipStream_t stream)
{
    const float* inputs = (const float*)d_in[0];
    const float* hx     = (const float*)d_in[1];
    const float* cx     = (const float*)d_in[2];
    const int*   matches= (const int*)  d_in[3];
    const float* W_ih   = (const float*)d_in[4];
    const float* W_hh   = (const float*)d_in[5];
    const float* b_ih   = (const float*)d_in[6];
    const float* b_hh   = (const float*)d_in[7];
    const float* W_out  = (const float*)d_in[8];
    const float* b_out  = (const float*)d_in[9];
    float* pmf = (float*)d_out;

    const int num_teams   = in_sizes[0] / 128;   // 100000
    const int num_matches = in_sizes[3] / 2;     // 1000000

    uint16_t* h_bf = (uint16_t*)d_ws;            // 100000*128*2B = 25.6 MB

    // Kernel 1: 8 hidden-unit tiles (y), grid-stride over M tiles (x)
    dim3 g1(512, 8, 1);
    lstm_cell_kernel<<<g1, 256, 0, stream>>>(inputs, hx, cx, W_ih, W_hh,
                                             b_ih, b_hh, h_bf, num_teams);

    // Kernel 2: 128 matches per block (8 waves x 16)
    int blocks2 = (num_matches + 127) / 128;
    match_head_kernel<<<blocks2, 256, 0, stream>>>(h_bf, matches, W_out, b_out,
                                                   pmf, num_matches);
}
